// GCN_49916109914694
// MI455X (gfx1250) — matrix operand order repaired and checked by reading, compiled, not run on hardware
//
#include <hip/hip_runtime.h>
#include <hip/hip_bf16.h>
#include <math.h>

// ---------------------------------------------------------------------------
// GCN forward for MI455X (gfx1250, wave32, WMMA).
// Layer-1 GEMM is HBM-bound streaming fp32 X (512MB -> ~22us floor at
// 23.3TB/s); f16 WMMA w/ f32 accumulation. Edge scatter is L2-resident
// (h/agg = 51MB each << 192MB L2). Weights pre-converted AND pre-transposed
// to f16 so the B tile is a pure byte copy -> async global->LDS via the
// gfx1250 builtin (ASYNCcnt path), float4 fallback otherwise.
// ---------------------------------------------------------------------------

typedef __attribute__((ext_vector_type(16))) _Float16 v16h;
typedef __attribute__((ext_vector_type(8)))  float    v8f;
typedef __attribute__((ext_vector_type(4)))  int      v4i;

#define HC   128
#define EPSC 1e-5f

#if defined(__gfx1250__) && __has_builtin(__builtin_amdgcn_global_load_async_to_lds_b128)
#define HAVE_ASYNC_LDS 1
typedef __attribute__((address_space(1))) v4i* gv4i_p;   // global
typedef __attribute__((address_space(3))) v4i* lv4i_p;   // LDS
#endif

// ---- fp32 -> f16 weight conversion + transpose: WT[n*K + k] = W[k*HC + n] ----
__global__ void k_w_to_f16T(const float* __restrict__ W, _Float16* __restrict__ WT, int K) {
  int idx = blockIdx.x * 256 + threadIdx.x;      // over HC*K, contiguous in k
  if (idx < K * HC) {
    int n = idx / K, k = idx % K;
    WT[idx] = (_Float16)W[(size_t)k * HC + n];
  }
}

// ---- degree / normalization ----
__global__ void k_deg_init(float* deg, int n) {
  int i = blockIdx.x * 256 + threadIdx.x;
  if (i < n) deg[i] = 1.0f;                      // self-loop contributes 1
}
__global__ void k_deg_edges(const int* __restrict__ dst, float* deg, int E) {
  int e = blockIdx.x * 256 + threadIdx.x;
  if (e < E) unsafeAtomicAdd(&deg[dst[e]], 1.0f);
}
__global__ void k_dinv(const float* __restrict__ deg, float* __restrict__ dinv, int n) {
  int i = blockIdx.x * 256 + threadIdx.x;
  if (i < n) { float d = deg[i]; dinv[i] = d > 0.f ? rsqrtf(d) : 0.f; }
}

// ---- WMMA GEMM: Hout[Nrows,128] = X[Nrows,K] (f32->f16) @ W[K,128] ----
// WT is the f16 TRANSPOSED weight: WT[n*K + k].
// Block = 256 thr (8 wave32) -> 128 rows x 128 cols per block.
__global__ __launch_bounds__(256) void k_gemm_wmma(const float* __restrict__ X,
                                                   const _Float16* __restrict__ WT,
                                                   float* __restrict__ Hout,
                                                   int Nrows, int K) {
  __shared__ _Float16 lA[128][40];   // A tile 128x32 f16 (+8 pad; 80B row = 16B aligned)
  __shared__ _Float16 lB[128][40];   // B tile, row n: k0..k0+31
  const int tid  = threadIdx.x;
  const int lane = tid & 31;
  const int wave = tid >> 5;
  const int mrow = lane & 15;
  const int g    = lane >> 4;        // lane-half per ISA 7.12.2 layouts
  const int blockRow = blockIdx.x * 128;

  v8f acc[8] = {};

  for (int k0 = 0; k0 < K; k0 += 32) {
    __syncthreads();
    // ---- B tile: pure 16B copies (512 units); async global->LDS if available.
    #pragma unroll
    for (int u = tid; u < 512; u += 256) {
      int n = u >> 2, seg = u & 3;                       // seg = 8-half chunk
      const _Float16* gsrc = WT + (size_t)n * K + k0 + seg * 8;
#ifdef HAVE_ASYNC_LDS
      __builtin_amdgcn_global_load_async_to_lds_b128(
          (gv4i_p)gsrc, (lv4i_p)&lB[n][seg * 8], 0, 0);
#else
      *(float4*)(&lB[n][seg * 8]) = *(const float4*)gsrc;
#endif
    }
    // ---- A tile: float4-coalesced loads, cvt f32->f16 into LDS.
    for (int q = tid; q < 1024; q += 256) {
      int row = q >> 3, kq = q & 7;
      int grow = blockRow + row;
      float4 val = make_float4(0.f, 0.f, 0.f, 0.f);
      if (grow < Nrows) {
        val = *(const float4*)(X + (size_t)grow * K + k0 + kq * 4);
        if (k0 + 32 < K)   // pull next K-tile toward WGP$/L2 (global_prefetch_b8)
          __builtin_prefetch(X + (size_t)grow * K + k0 + 32 + kq * 4, 0, 0);
      }
      _Float16* p = &lA[row][kq * 4];
      p[0] = (_Float16)val.x; p[1] = (_Float16)val.y;
      p[2] = (_Float16)val.z; p[3] = (_Float16)val.w;
    }
#ifdef HAVE_ASYNC_LDS
#if __has_builtin(__builtin_amdgcn_s_wait_asynccnt)
    __builtin_amdgcn_s_wait_asynccnt(0);
#else
    asm volatile("s_wait_asynccnt 0" ::: "memory");
#endif
#endif
    __syncthreads();

    // A fragment: lane=M; VGPR v holds K pair {2v,2v+1} (+8 per lane-half, +16 for v>=4)
    v16h afrag;
    {
      const _Float16* arow = &lA[wave * 16 + mrow][0];
      #pragma unroll
      for (int v = 0; v < 8; ++v) {
        int kb = (v < 4 ? 2 * v : 16 + 2 * (v - 4)) + g * 8;
        afrag[2 * v]     = arow[kb];
        afrag[2 * v + 1] = arow[kb + 1];
      }
    }
    // Preload ALL 8 B fragments, then issue the 8 WMMAs back-to-back so the
    // XDL pipe isn't round-tripping on per-tile ds_load + s_wait_dscnt.
    v16h bf[8];
    #pragma unroll
    for (int nt = 0; nt < 8; ++nt) {
      const _Float16* brow = &lB[nt * 16 + mrow][g * 8];
      #pragma unroll
      for (int j = 0; j < 16; ++j) bf[nt][j] = brow[j < 8 ? j : j + 8];
    }
    #pragma unroll
    for (int nt = 0; nt < 8; ++nt)
      acc[nt] = __builtin_amdgcn_wmma_f32_16x16x32_f16(
          false, afrag, false, bf[nt], (short)0, acc[nt], false, false);
  }

  // D layout: VGPR r -> row (r + 8*g), col = lane%16.
  #pragma unroll
  for (int nt = 0; nt < 8; ++nt) {
    #pragma unroll
    for (int r = 0; r < 8; ++r) {
      int row = blockRow + wave * 16 + r + 8 * g;
      if (row < Nrows) Hout[(size_t)row * HC + nt * 16 + mrow] = acc[nt][r];
    }
  }
}

// ---- agg = bias + dinv^2 * h (self-loop term) ----
__global__ void k_init_agg(const float* __restrict__ h, const float* __restrict__ dinv,
                           const float* __restrict__ bias, float* __restrict__ agg, int n) {
  int idx = blockIdx.x * 256 + threadIdx.x;
  if (idx < n) {
    int i = idx >> 7, c = idx & 127;
    float di = dinv[i];
    agg[idx] = bias[c] + h[idx] * di * di;
  }
}
// ---- edge scatter: 8 edges per block, lane covers 4 channels (float4 gather) ----
__global__ __launch_bounds__(256) void k_edge_scatter(
    const float* __restrict__ h, const int* __restrict__ src,
    const int* __restrict__ dst, const float* __restrict__ dinv,
    float* __restrict__ agg, int E) {
  int e = blockIdx.x * 8 + (threadIdx.x >> 5);
  if (e >= E) return;
  int c4 = (threadIdx.x & 31) * 4;
  int s = src[e], d = dst[e];
  float w = dinv[s] * dinv[d];
  const float4 hv = *(const float4*)(h + (size_t)s * HC + c4);
  float* ap = agg + (size_t)d * HC + c4;
  unsafeAtomicAdd(ap + 0, hv.x * w);
  unsafeAtomicAdd(ap + 1, hv.y * w);
  unsafeAtomicAdd(ap + 2, hv.z * w);
  unsafeAtomicAdd(ap + 3, hv.w * w);
}

// ---- eval-mode BN (optionally after ReLU) ----
__global__ void k_bn(const float* __restrict__ in, float* __restrict__ out,
                     const float* __restrict__ g, const float* __restrict__ be,
                     const float* __restrict__ m, const float* __restrict__ v,
                     int n, int relu) {
  int idx = blockIdx.x * 256 + threadIdx.x;
  if (idx < n) {
    int c = idx & 127;
    float t = in[idx];
    if (relu) t = fmaxf(t, 0.f);
    out[idx] = (t - m[c]) * rsqrtf(v[c] + EPSC) * g[c] + be[c];
  }
}

// ---- global mean pool (atomic sums + counts) ----
__global__ void k_pool(const float* __restrict__ x, const int* __restrict__ batch,
                       float* __restrict__ pooled, float* __restrict__ cnt, int n) {
  int idx = blockIdx.x * 256 + threadIdx.x;
  if (idx < n) {
    int i = idx >> 7, c = idx & 127;
    int b = batch[i];
    unsafeAtomicAdd(&pooled[(size_t)b * HC + c], x[idx]);
    if (c == 0) unsafeAtomicAdd(&cnt[b], 1.0f);
  }
}

// ---- z = pooled/cnt + x_seq@fcW + fcb ; heads: sigmoid(z@l0W), sigmoid(z@l1W) x2 ----
__global__ __launch_bounds__(128) void k_head(
    const float* __restrict__ pooled, const float* __restrict__ cnt,
    const float* __restrict__ x_seq, const float* __restrict__ fcW,
    const float* __restrict__ fcb,
    const float* __restrict__ l0W, const float* __restrict__ l0b,
    const float* __restrict__ l1W, const float* __restrict__ l1b,
    float* __restrict__ out, int F_IN, int C0c, int C1c) {
  __shared__ float z[HC];
  int b = blockIdx.x, t = threadIdx.x;
  float acc = fcb[t];
  for (int k = 0; k < F_IN; ++k)
    acc += x_seq[(size_t)b * F_IN + k] * fcW[(size_t)k * HC + t];
  float c = fmaxf(cnt[b], 1.0f);
  z[t] = pooled[(size_t)b * HC + t] / c + acc;
  __syncthreads();
  int total = C0c + C1c;
  size_t obase = (size_t)b * (C0c + 2 * C1c);
  for (int j = t; j < total; j += 128) {
    if (j < C0c) {
      float s = l0b[j];
      for (int k = 0; k < HC; ++k) s += z[k] * l0W[(size_t)k * C0c + j];
      out[obase + j] = 1.f / (1.f + __expf(-s));
    } else {
      int j2 = j - C0c;
      float s = l1b[j2];
      for (int k = 0; k < HC; ++k) s += z[k] * l1W[(size_t)k * C1c + j2];
      float sv = 1.f / (1.f + __expf(-s));
      out[obase + C0c + j2] = sv;
      out[obase + C0c + C1c + j2] = sv;      // head-2 emitted twice
    }
  }
}

extern "C" void kernel_launch(void* const* d_in, const int* in_sizes, int n_in,
                              void* d_out, int out_size, void* d_ws, size_t ws_size,
                              hipStream_t stream) {
  const float* x_res = (const float*)d_in[0];
  const int*   ei    = (const int*)d_in[1];
  const int*   batch = (const int*)d_in[2];
  const float* x_seq = (const float*)d_in[3];
  const float* W1 = (const float*)d_in[4];  const float* b1 = (const float*)d_in[5];
  const float* W2 = (const float*)d_in[6];  const float* b2 = (const float*)d_in[7];
  const float* W3 = (const float*)d_in[8];  const float* b3 = (const float*)d_in[9];
  const float* g1 = (const float*)d_in[10]; const float* be1 = (const float*)d_in[11];
  const float* m1 = (const float*)d_in[12]; const float* v1  = (const float*)d_in[13];
  const float* g2 = (const float*)d_in[14]; const float* be2 = (const float*)d_in[15];
  const float* m2 = (const float*)d_in[16]; const float* v2  = (const float*)d_in[17];
  const float* g3 = (const float*)d_in[18]; const float* be3 = (const float*)d_in[19];
  const float* m3 = (const float*)d_in[20]; const float* v3  = (const float*)d_in[21];
  const float* fcW = (const float*)d_in[22]; const float* fcb = (const float*)d_in[23];
  const float* l0W = (const float*)d_in[24]; const float* l0b = (const float*)d_in[25];
  const float* l1W = (const float*)d_in[26]; const float* l1b = (const float*)d_in[27];
  float* out = (float*)d_out;

  const int E    = in_sizes[1] / 2;
  const int N    = in_sizes[2];
  const int F_IN = in_sizes[0] / N;
  const int B    = in_sizes[3] / F_IN;
  const int C0   = in_sizes[25];
  const int C1   = in_sizes[27];
  const int NH   = N * HC;

  const int* e_src = ei;
  const int* e_dst = ei + E;

  // workspace carve-out (256B aligned); ~104 MB total
  char* ws = (char*)d_ws; size_t off = 0;
  auto carve = [&](size_t bytes) -> void* {
    void* p = ws + off; off += (bytes + 255) & ~(size_t)255; return p;
  };
  float*     bufA   = (float*)carve((size_t)NH * 4);
  float*     bufB   = (float*)carve((size_t)NH * 4);
  float*     deg    = (float*)carve((size_t)N * 4);
  float*     dinv   = (float*)carve((size_t)N * 4);
  float*     pooled = (float*)carve((size_t)B * HC * 4);
  float*     cnt    = (float*)carve((size_t)B * 4);
  _Float16*  w1t    = (_Float16*)carve((size_t)F_IN * HC * 2);
  _Float16*  w2t    = (_Float16*)carve((size_t)HC * HC * 2);
  _Float16*  w3t    = (_Float16*)carve((size_t)HC * HC * 2);

  dim3 blk(256);
  auto nb = [](int n) { return dim3((unsigned)((n + 255) / 256)); };
  dim3 gemm_grid((unsigned)((N + 127) / 128));
  dim3 edge_grid((unsigned)((E + 7) / 8));

  // weights -> f16, transposed [n][k]
  k_w_to_f16T<<<nb(F_IN * HC), blk, 0, stream>>>(W1, w1t, F_IN);
  k_w_to_f16T<<<nb(HC * HC),   blk, 0, stream>>>(W2, w2t, HC);
  k_w_to_f16T<<<nb(HC * HC),   blk, 0, stream>>>(W3, w3t, HC);

  // symmetric normalization (self-loops included)
  k_deg_init <<<nb(N), blk, 0, stream>>>(deg, N);
  k_deg_edges<<<nb(E), blk, 0, stream>>>(e_dst, deg, E);
  k_dinv     <<<nb(N), blk, 0, stream>>>(deg, dinv, N);

  // ---- layer 1: x_res @ W1 -> bufA; scatter -> bufB; ReLU+BN -> bufA
  k_gemm_wmma   <<<gemm_grid, blk, 0, stream>>>(x_res, w1t, bufA, N, F_IN);
  k_init_agg    <<<nb(NH), blk, 0, stream>>>(bufA, dinv, b1, bufB, NH);
  k_edge_scatter<<<edge_grid, blk, 0, stream>>>(bufA, e_src, e_dst, dinv, bufB, E);
  k_bn          <<<nb(NH), blk, 0, stream>>>(bufB, bufA, g1, be1, m1, v1, NH, 1);

  // ---- layer 2: bufA @ W2 -> bufB; scatter -> bufA; ReLU+BN -> bufB
  k_gemm_wmma   <<<gemm_grid, blk, 0, stream>>>(bufA, w2t, bufB, N, HC);
  k_init_agg    <<<nb(NH), blk, 0, stream>>>(bufB, dinv, b2, bufA, NH);
  k_edge_scatter<<<edge_grid, blk, 0, stream>>>(bufB, e_src, e_dst, dinv, bufA, E);
  k_bn          <<<nb(NH), blk, 0, stream>>>(bufA, bufB, g2, be2, m2, v2, NH, 1);

  // ---- layer 3: bufB @ W3 -> bufA; scatter -> bufB; BN (no ReLU) -> bufA
  k_gemm_wmma   <<<gemm_grid, blk, 0, stream>>>(bufB, w3t, bufA, N, HC);
  k_init_agg    <<<nb(NH), blk, 0, stream>>>(bufA, dinv, b3, bufB, NH);
  k_edge_scatter<<<edge_grid, blk, 0, stream>>>(bufA, e_src, e_dst, dinv, bufB, E);
  k_bn          <<<nb(NH), blk, 0, stream>>>(bufB, bufA, g3, be3, m3, v3, NH, 0);

  // ---- pool + heads
  (void)hipMemsetAsync(pooled, 0, (size_t)B * HC * 4, stream);
  (void)hipMemsetAsync(cnt,    0, (size_t)B * 4,      stream);
  k_pool<<<nb(NH), blk, 0, stream>>>(bufA, batch, pooled, cnt, NH);
  k_head<<<dim3((unsigned)B), dim3(128), 0, stream>>>(
      pooled, cnt, x_seq, fcW, fcb, l0W, l0b, l1W, l1b, out, F_IN, C0, C1);
}
